// MambaLayer_41626823033161
// MI455X (gfx1250) — matrix operand order repaired and checked by reading, not compiled, not run on hardware
//
#include <hip/hip_runtime.h>
#include <hip/hip_bf16.h>
#include <math.h>

// ---------------- problem constants ----------------
#define Bb   2
#define Ll   2048
#define Ee   1024
#define DIN  2048
#define DST  16
#define DCV  4
#define DTR  64
#define RPROJ (DTR + 2*DST)        // 96
#define MTOK (Bb*Ll)               // 4096
#define NCH  64                    // scan chunks along L
#define CLEN (Ll/NCH)              // 32 steps per chunk

typedef _Float16 h16;
typedef __attribute__((ext_vector_type(16))) _Float16 v16h;
typedef __attribute__((ext_vector_type(8)))  _Float16 v8h;
typedef __attribute__((ext_vector_type(8)))  float    v8f;

// ---------------- WMMA fragment loaders (gfx1250 16-bit layouts) ----------------
// A-matrix 16x32 (MxK): lane<16 -> M=lane, K in {0..7, 16..23};
//                       lane>=16 -> M=lane-16, K in {8..15, 24..31}
__device__ __forceinline__ v16h load_a_frag(const h16* __restrict__ a_row, int kb, bool hi) {
  union { v16h v; v8h h[2]; } u;
  const int ko = hi ? 8 : 0;
  u.h[0] = *(const v8h*)(a_row + kb + ko);
  u.h[1] = *(const v8h*)(a_row + kb + 16 + ko);
  return u.v;
}
// B-matrix 32x16 (KxN), sourced from row-major W (N x K):
// lane<16 -> N=lane, K in {0..7, 16..23}; lane>=16 -> N=lane-16, K in {8..15, 24..31}
__device__ __forceinline__ v16h load_b_frag(const h16* __restrict__ b_row, int kb, bool hi) {
  union { v16h v; v8h h[2]; } u;
  const int ko = hi ? 8 : 0;
  u.h[0] = *(const v8h*)(b_row + kb + ko);
  u.h[1] = *(const v8h*)(b_row + kb + 16 + ko);
  return u.v;
}

// ---------------- generic WMMA GEMM: C[M,N] = A[M,K] * B[N,K]^T (+resid) ----------------
// One wave computes a (16*TM) x (16*TN) tile. 4 waves / block.
template<int TM, int TN, bool RESID>
__global__ __launch_bounds__(128) void gemm_wmma_kern(
    const h16* __restrict__ A, const h16* __restrict__ Bw,
    const float* __restrict__ resid, float* __restrict__ C,
    int M, int N, int K) {
  const int lane = threadIdx.x & 31;
  const int wave = threadIdx.x >> 5;
  const int tiles_n = N / (16*TN);
  const int tile = blockIdx.x * 4 + wave;
  const int tm = (tile / tiles_n) * (16*TM);
  const int tn = (tile % tiles_n) * (16*TN);
  const bool hi = (lane & 16) != 0;
  const int lr = lane & 15;

  const h16* arow[TM];
#pragma unroll
  for (int i = 0; i < TM; ++i) arow[i] = A + (size_t)(tm + i*16 + lr) * (size_t)K;
  const h16* brow[TN];
#pragma unroll
  for (int j = 0; j < TN; ++j) brow[j] = Bw + (size_t)(tn + j*16 + lr) * (size_t)K;

  v8f acc[TM][TN] = {};

  for (int kb = 0; kb < K; kb += 32) {
    v16h af[TM], bf[TN];
#pragma unroll
    for (int i = 0; i < TM; ++i) af[i] = load_a_frag(arow[i], kb, hi);
#pragma unroll
    for (int j = 0; j < TN; ++j) bf[j] = load_b_frag(brow[j], kb, hi);
#pragma unroll
    for (int i = 0; i < TM; ++i)
#pragma unroll
      for (int j = 0; j < TN; ++j)
        acc[i][j] = __builtin_amdgcn_wmma_f32_16x16x32_f16(
            false, af[i], false, bf[j], (short)0, acc[i][j], false, false);
  }

  // C/D layout: VGPR r -> row (hi?8:0)+r, col = lane&15
  const int rbase = hi ? 8 : 0;
#pragma unroll
  for (int i = 0; i < TM; ++i) {
#pragma unroll
    for (int rr = 0; rr < 8; ++rr) {
      const int row = tm + i*16 + rbase + rr;
      float* crow = C + (size_t)row * (size_t)N + tn;
      const float* xrow = RESID ? (resid + (size_t)row * (size_t)N + tn) : nullptr;
#pragma unroll
      for (int j = 0; j < TN; ++j) {
        float v = acc[i][j][rr];
        if (RESID) v += xrow[j*16 + lr];
        crow[j*16 + lr] = v;
      }
    }
  }
}

// ---------------- layernorm (+f16 convert), one block per token row ----------------
__global__ __launch_bounds__(256) void ln_kern(const float* __restrict__ x,
    const float* __restrict__ g, const float* __restrict__ b,
    h16* __restrict__ xn) {
  __shared__ float red[256];
  const int m = blockIdx.x;
  const float* xr = x + (size_t)m * Ee;
  float v[4]; float s = 0.f;
#pragma unroll
  for (int i = 0; i < 4; ++i) { v[i] = xr[threadIdx.x + i*256]; s += v[i]; }
  red[threadIdx.x] = s; __syncthreads();
  for (int o = 128; o > 0; o >>= 1) { if (threadIdx.x < o) red[threadIdx.x] += red[threadIdx.x + o]; __syncthreads(); }
  const float mu = red[0] * (1.0f / Ee);
  __syncthreads();
  float s2 = 0.f;
#pragma unroll
  for (int i = 0; i < 4; ++i) { float d = v[i] - mu; s2 += d*d; }
  red[threadIdx.x] = s2; __syncthreads();
  for (int o = 128; o > 0; o >>= 1) { if (threadIdx.x < o) red[threadIdx.x] += red[threadIdx.x + o]; __syncthreads(); }
  const float rs = rsqrtf(red[0] * (1.0f / Ee) + 1e-5f);
#pragma unroll
  for (int i = 0; i < 4; ++i) {
    const int e = threadIdx.x + i*256;
    xn[(size_t)m * Ee + e] = (h16)((v[i] - mu) * rs * g[e] + b[e]);
  }
}

// ---------------- f32 -> f16 convert ----------------
__global__ __launch_bounds__(256) void cvt_f2h_kern(const float* __restrict__ src,
                                                    h16* __restrict__ dst, int n) {
  const int i = blockIdx.x * 256 + threadIdx.x;
  if (i < n) dst[i] = (h16)src[i];
}

// ---------------- causal depthwise conv + SiLU ----------------
__global__ __launch_bounds__(256) void conv_silu_kern(const float* __restrict__ xz,
    const float* __restrict__ cw, const float* __restrict__ cb,
    float* __restrict__ u_f, h16* __restrict__ u_h) {
  const int idx = blockIdx.x * 256 + threadIdx.x;       // over B*L*DIN
  if (idx >= Bb*Ll*DIN) return;
  const int d = idx % DIN;
  const int l = (idx / DIN) % Ll;
  const int b = idx / (DIN * Ll);
  float acc = cb[d];
#pragma unroll
  for (int k = 0; k < DCV; ++k) {
    const int ls = l + k - (DCV - 1);
    if (ls >= 0)
      acc += cw[d*DCV + k] * xz[(size_t)(b*Ll + ls) * (2*DIN) + d];
  }
  const float s = acc / (1.0f + __expf(-acc));
  u_f[idx] = s;
  u_h[idx] = (h16)s;
}

// ---------------- dt_in slice of dbc -> f16 ----------------
__global__ __launch_bounds__(256) void dtin_cvt_kern(const float* __restrict__ dbc,
                                                     h16* __restrict__ dtin) {
  const int idx = blockIdx.x * 256 + threadIdx.x;       // over MTOK*DTR
  if (idx >= MTOK*DTR) return;
  const int m = idx / DTR, r = idx % DTR;
  dtin[idx] = (h16)dbc[(size_t)m * RPROJ + r];
}

// ---------------- softplus(dt_pre + dt_bias) in place ----------------
__global__ __launch_bounds__(256) void softplus_kern(float* __restrict__ dt,
                                                     const float* __restrict__ bias) {
  const int idx = blockIdx.x * 256 + threadIdx.x;       // over MTOK*DIN
  if (idx >= MTOK*DIN) return;
  const float v = dt[idx] + bias[idx % DIN];
  dt[idx] = (v > 20.f) ? v : log1pf(__expf(v));
}

// ================= chunked parallel selective scan =================
// Recurrence per (b,d,n): h <- exp(dt*a[n])*h + dt*u*B[l,n]; y = sum_n h*C[l,n].
// Chunked: pass1 computes per-chunk (P = exp(a*sum dt), S = local end state from 0),
// pass2 combines chunks sequentially (64 steps) and stores chunk-start states,
// pass3 replays each chunk from its start state and emits y.

// pass 1: parallel over Bb*NCH*DIN
__global__ __launch_bounds__(256) void scan_p1_kern(const float* __restrict__ dt,
    const float* __restrict__ u_f, const float* __restrict__ dbc,
    const float* __restrict__ A_log, float* __restrict__ P, float* __restrict__ S) {
  const int t = blockIdx.x * 256 + threadIdx.x;
  if (t >= Bb*NCH*DIN) return;
  const int d = t % DIN;
  const int c = (t / DIN) % NCH;
  const int b = t / (DIN * NCH);
  float a[DST], s[DST];
#pragma unroll
  for (int n = 0; n < DST; ++n) { a[n] = -__expf(A_log[d*DST + n]); s[n] = 0.f; }
  float sdt = 0.f;
  for (int i = 0; i < CLEN; ++i) {
    const size_t row = (size_t)(b*Ll + c*CLEN + i);
    const float dtv = dt[row*DIN + d];
    const float dtu = dtv * u_f[row*DIN + d];
    const float* bc = dbc + row * RPROJ;
    sdt += dtv;
#pragma unroll
    for (int n = 0; n < DST; ++n) {
      const float da = __expf(dtv * a[n]);
      s[n] = da * s[n] + dtu * bc[DTR + n];
    }
  }
  const size_t base = (size_t)(b*NCH + c) * DST * DIN + d;
#pragma unroll
  for (int n = 0; n < DST; ++n) {
    P[base + (size_t)n*DIN] = __expf(a[n] * sdt);   // product of das over chunk
    S[base + (size_t)n*DIN] = s[n];
  }
}

// pass 2: parallel over Bb*DIN; sequential over the 64 chunks.
// Overwrites P[c] with the chunk-START state (P fully rewritten by pass1 each call).
__global__ __launch_bounds__(256) void scan_p2_kern(float* __restrict__ P,
                                                    const float* __restrict__ S) {
  const int t = blockIdx.x * 256 + threadIdx.x;
  if (t >= Bb*DIN) return;
  const int d = t % DIN, b = t / DIN;
  float h[DST];
#pragma unroll
  for (int n = 0; n < DST; ++n) h[n] = 0.f;
  for (int c = 0; c < NCH; ++c) {
    const size_t base = (size_t)(b*NCH + c) * DST * DIN + d;
#pragma unroll
    for (int n = 0; n < DST; ++n) {
      const size_t ix = base + (size_t)n*DIN;
      const float p = P[ix];
      const float s = S[ix];
      P[ix] = h[n];                 // store start state for this chunk
      h[n] = p * h[n] + s;
    }
  }
}

// pass 3: parallel over Bb*NCH*DIN; replay chunk from start state, emit y.
__global__ __launch_bounds__(256) void scan_p3_kern(const float* __restrict__ dt,
    const float* __restrict__ u_f, const float* __restrict__ dbc,
    const float* __restrict__ A_log, const float* __restrict__ H0,
    float* __restrict__ y) {
  const int t = blockIdx.x * 256 + threadIdx.x;
  if (t >= Bb*NCH*DIN) return;
  const int d = t % DIN;
  const int c = (t / DIN) % NCH;
  const int b = t / (DIN * NCH);
  float a[DST], h[DST];
  const size_t base = (size_t)(b*NCH + c) * DST * DIN + d;
#pragma unroll
  for (int n = 0; n < DST; ++n) {
    a[n] = -__expf(A_log[d*DST + n]);
    h[n] = H0[base + (size_t)n*DIN];
  }
  for (int i = 0; i < CLEN; ++i) {
    const size_t row = (size_t)(b*Ll + c*CLEN + i);
    const float dtv = dt[row*DIN + d];
    const float dtu = dtv * u_f[row*DIN + d];
    const float* bc = dbc + row * RPROJ;
    float yv = 0.f;
#pragma unroll
    for (int n = 0; n < DST; ++n) {
      const float da = __expf(dtv * a[n]);
      h[n] = da * h[n] + dtu * bc[DTR + n];
      yv += h[n] * bc[DTR + DST + n];
    }
    y[row*DIN + d] = yv;
  }
}

// ---------------- gating: (y + u*D) * silu(z) -> f16 ----------------
__global__ __launch_bounds__(256) void gate_kern(const float* __restrict__ y,
    const float* __restrict__ u_f, const float* __restrict__ Dv,
    const float* __restrict__ xz, h16* __restrict__ y_h) {
  const int idx = blockIdx.x * 256 + threadIdx.x;       // over MTOK*DIN
  if (idx >= MTOK*DIN) return;
  const int d = idx % DIN;
  const size_t row = (size_t)(idx / DIN);
  const float z = xz[row*(2*DIN) + DIN + d];
  const float g = z / (1.0f + __expf(-z));
  y_h[idx] = (h16)((y[idx] + u_f[idx]*Dv[d]) * g);
}

// ---------------- workspace layout (all sizes are multiples of 256B) ----------------
#define SZ_WI   ((size_t)(2*DIN)*Ee*2)        // W_in  f16: 4096x1024
#define SZ_WX   ((size_t)RPROJ*DIN*2)         // W_x   f16: 96x2048
#define SZ_WDT  ((size_t)DIN*DTR*2)           // W_dt  f16: 2048x64
#define SZ_WO   ((size_t)Ee*DIN*2)            // W_out f16: 1024x2048
#define SZ_XN   ((size_t)MTOK*Ee*2)           // xn    f16
#define SZ_XZ   ((size_t)MTOK*(2*DIN)*4)      // xz    f32
#define SZ_UF   ((size_t)MTOK*DIN*4)          // u     f32
#define SZ_UH   ((size_t)MTOK*DIN*2)          // u     f16
#define SZ_DBC  ((size_t)MTOK*RPROJ*4)        // dbc   f32
#define SZ_DTIN ((size_t)MTOK*DTR*2)          // dt_in f16
#define SZ_DT   ((size_t)MTOK*DIN*4)          // dt    f32
#define SZ_Y    ((size_t)MTOK*DIN*4)          // y     f32
#define SZ_YH   ((size_t)MTOK*DIN*2)          // y     f16
#define SZ_SP   ((size_t)Bb*NCH*DIN*DST*4)    // scan P / chunk-start states
#define SZ_SS   ((size_t)Bb*NCH*DIN*DST*4)    // scan S (local end states)

extern "C" void kernel_launch(void* const* d_in, const int* in_sizes, int n_in,
                              void* d_out, int out_size, void* d_ws, size_t ws_size,
                              hipStream_t stream) {
  const float* x      = (const float*)d_in[0];
  const float* ln_g   = (const float*)d_in[1];
  const float* ln_b   = (const float*)d_in[2];
  const float* W_in   = (const float*)d_in[3];
  const float* conv_w = (const float*)d_in[4];
  const float* conv_b = (const float*)d_in[5];
  const float* W_x    = (const float*)d_in[6];
  const float* W_dt   = (const float*)d_in[7];
  const float* dt_bias= (const float*)d_in[8];
  const float* A_log  = (const float*)d_in[9];
  const float* Dvec   = (const float*)d_in[10];
  const float* W_out  = (const float*)d_in[11];
  float* out = (float*)d_out;

  char* ws = (char*)d_ws;
  size_t off = 0;
  h16*   Wi   = (h16*)(ws + off);  off += SZ_WI;
  h16*   Wx   = (h16*)(ws + off);  off += SZ_WX;
  h16*   Wdt  = (h16*)(ws + off);  off += SZ_WDT;
  h16*   Wo   = (h16*)(ws + off);  off += SZ_WO;
  h16*   XN   = (h16*)(ws + off);  off += SZ_XN;
  float* XZ   = (float*)(ws + off); off += SZ_XZ;
  float* UF   = (float*)(ws + off); off += SZ_UF;
  h16*   UH   = (h16*)(ws + off);  off += SZ_UH;
  float* DBC  = (float*)(ws + off); off += SZ_DBC;
  h16*   DTIN = (h16*)(ws + off);  off += SZ_DTIN;
  float* DT   = (float*)(ws + off); off += SZ_DT;
  float* Yf   = (float*)(ws + off); off += SZ_Y;
  h16*   Yh   = (h16*)(ws + off);  off += SZ_YH;
  float* SP   = (float*)(ws + off); off += SZ_SP;
  float* SS   = (float*)(ws + off); off += SZ_SS;

  // 1) weights -> f16
  { int n = 2*DIN*Ee;   cvt_f2h_kern<<<(n+255)/256, 256, 0, stream>>>(W_in,  Wi,  n); }
  { int n = RPROJ*DIN;  cvt_f2h_kern<<<(n+255)/256, 256, 0, stream>>>(W_x,   Wx,  n); }
  { int n = DIN*DTR;    cvt_f2h_kern<<<(n+255)/256, 256, 0, stream>>>(W_dt,  Wdt, n); }
  { int n = Ee*DIN;     cvt_f2h_kern<<<(n+255)/256, 256, 0, stream>>>(W_out, Wo,  n); }

  // 2) layernorm + f16
  ln_kern<<<MTOK, 256, 0, stream>>>(x, ln_g, ln_b, XN);

  // 3) xz = xn @ W_in^T   (M=4096, N=4096, K=1024)
  { int tiles = (MTOK/64) * ((2*DIN)/64);
    gemm_wmma_kern<4,4,false><<<tiles/4, 128, 0, stream>>>(XN, Wi, nullptr, XZ, MTOK, 2*DIN, Ee); }

  // 4) depthwise causal conv + SiLU
  { int n = Bb*Ll*DIN; conv_silu_kern<<<(n+255)/256, 256, 0, stream>>>(XZ, conv_w, conv_b, UF, UH); }

  // 5) dbc = u @ W_x^T    (M=4096, N=96, K=2048)
  { int tiles = (MTOK/64) * (RPROJ/32);
    gemm_wmma_kern<4,2,false><<<tiles/4, 128, 0, stream>>>(UH, Wx, nullptr, DBC, MTOK, RPROJ, DIN); }

  // 6) dt_in slice -> f16
  { int n = MTOK*DTR; dtin_cvt_kern<<<(n+255)/256, 256, 0, stream>>>(DBC, DTIN); }

  // 7) dt_pre = dt_in @ W_dt^T  (M=4096, N=2048, K=64)
  { int tiles = (MTOK/64) * (DIN/64);
    gemm_wmma_kern<4,4,false><<<tiles/4, 128, 0, stream>>>(DTIN, Wdt, nullptr, DT, MTOK, DIN, DTR); }

  // 8) dt = softplus(dt_pre + dt_bias)
  { int n = MTOK*DIN; softplus_kern<<<(n+255)/256, 256, 0, stream>>>(DT, dt_bias); }

  // 9) chunked selective scan (pass1 parallel, pass2 combine, pass3 replay+emit)
  { int n = Bb*NCH*DIN; scan_p1_kern<<<(n+255)/256, 256, 0, stream>>>(DT, UF, DBC, A_log, SP, SS); }
  { int n = Bb*DIN;     scan_p2_kern<<<(n+255)/256, 256, 0, stream>>>(SP, SS); }
  { int n = Bb*NCH*DIN; scan_p3_kern<<<(n+255)/256, 256, 0, stream>>>(DT, UF, DBC, A_log, SP, Yf); }

  // 10) gating -> f16
  { int n = MTOK*DIN; gate_kern<<<(n+255)/256, 256, 0, stream>>>(Yf, UF, Dvec, XZ, Yh); }

  // 11) out = x + y @ W_out^T   (M=4096, N=1024, K=2048)
  { int tiles = (MTOK/64) * (Ee/64);
    gemm_wmma_kern<4,4,true><<<tiles/4, 128, 0, stream>>>(Yh, Wo, x, out, MTOK, Ee, DIN); }

  (void)in_sizes; (void)n_in; (void)out_size; (void)ws_size;
}
